// Attention_56427280335498
// MI455X (gfx1250) — matrix operand order repaired and checked by reading, not compiled, not run on hardware
//
#include <hip/hip_runtime.h>

#define Bb 4
#define S  1024
#define D  512
#define H  8
#define DH 64

typedef __bf16 bf16;
typedef __attribute__((ext_vector_type(16))) __bf16 v16bf;
typedef __attribute__((ext_vector_type(8)))  __bf16 v8bf;
typedef __attribute__((ext_vector_type(8)))  float  v8f;

__device__ inline bf16 to_bf16(float f) {
    union { float f; unsigned u; } a; a.f = f;
    unsigned u = a.u + 0x7FFFu + ((a.u >> 16) & 1u);  // RNE
    union { unsigned short s; bf16 b; } o; o.s = (unsigned short)(u >> 16);
    return o.b;
}

// LDS byte offset of a __shared__ object: generic LDS pointers carry the
// offset in addr[31:0] (aperture bits live in addr[63:32] per CDNA5 ISA).
__device__ inline unsigned lds_off(const void* p) {
    return (unsigned)(uintptr_t)p;
}

// Async global -> LDS copy of 32 contiguous bytes (2 x b128), ASYNCcnt-tracked.
__device__ inline void async_copy32(unsigned lds_addr, const void* gptr) {
    unsigned long long ga = (unsigned long long)(uintptr_t)gptr;
    asm volatile("global_load_async_to_lds_b128 %0, %1, off\n\t"
                 "global_load_async_to_lds_b128 %0, %1, off offset:16"
                 :: "v"(lds_addr), "v"(ga) : "memory");
}
__device__ inline void async_wait0() {
    asm volatile("s_wait_asynccnt 0" ::: "memory");
}

// A-fragment (16x32, M x K, 16-bit): lane L holds row M=L%16,
// K chunks at {hi*8 .. hi*8+7} and {hi*8+16 .. hi*8+23}, hi = L/16.
__device__ inline v16bf load_frag_rows(const bf16* base, int rowStride, int lane) {
    int r = lane & 15, hi = lane >> 4;
    const bf16* p = base + (size_t)r * rowStride + hi * 8;
    v8bf lo = *(const v8bf*)(p);
    v8bf hh = *(const v8bf*)(p + 16);
    v16bf out;
#pragma unroll
    for (int i = 0; i < 8; i++) { out[i] = lo[i]; out[8 + i] = hh[i]; }
    return out;
}

// B-fragment (32x16, K x N, 16-bit): lane L holds column N=L%16,
// K chunks at {hi*8 .. hi*8+7} and {hi*8+16 .. hi*8+23}. Source stored [N][K] row-major.
__device__ inline v16bf load_frag_cols(const bf16* base, int rowStride, int lane) {
    int n = lane & 15, hi = lane >> 4;
    const bf16* p = base + (size_t)n * rowStride + hi * 8;
    v8bf lo = *(const v8bf*)(p);
    v8bf hh = *(const v8bf*)(p + 16);
    v16bf out;
#pragma unroll
    for (int i = 0; i < 8; i++) { out[i] = lo[i]; out[8 + i] = hh[i]; }
    return out;
}

// ---------------- LayerNorm (per row of D=512) + bf16 convert ----------------
__global__ void ln_kernel(const float* __restrict__ x, const float* __restrict__ g,
                          const float* __restrict__ bvec, bf16* __restrict__ out) {
    int row = blockIdx.x, t = threadIdx.x;
    const float* xr = x + (size_t)row * D;
    float x0 = xr[t], x1 = xr[t + 256];
    float s = x0 + x1, q = x0 * x0 + x1 * x1;
#pragma unroll
    for (int o = 16; o > 0; o >>= 1) { s += __shfl_xor(s, o, 32); q += __shfl_xor(q, o, 32); }
    __shared__ float ss[8], sq[8];
    int w = t >> 5, l = t & 31;
    if (l == 0) { ss[w] = s; sq[w] = q; }
    __syncthreads();
    if (t == 0) {
        float S1 = 0.f, Q1 = 0.f;
        for (int i = 0; i < 8; i++) { S1 += ss[i]; Q1 += sq[i]; }
        ss[0] = S1; sq[0] = Q1;
    }
    __syncthreads();
    float mean = ss[0] / (float)D;
    float var  = sq[0] / (float)D - mean * mean;
    float rs = rsqrtf(var + 1e-5f);
    bf16* orow = out + (size_t)row * D;
    orow[t]       = to_bf16((x0 - mean) * rs * g[t] + bvec[t]);
    orow[t + 256] = to_bf16((x1 - mean) * rs * g[t + 256] + bvec[t + 256]);
}

// ---------------- f32 -> bf16 elementwise ----------------
__global__ void cvt_kernel(const float* __restrict__ in, bf16* __restrict__ out, int n) {
    int i = blockIdx.x * 256 + threadIdx.x;
    if (i < n) out[i] = to_bf16(in[i]);
}

// ---------------- Tiled bf16 WMMA GEMM: C[M,N] = A[M,K] * B[K,N] (+bias) ----
// Block tile 128x64, BK=32, 8 waves. Compile-time epilogue selection.
template <bool HASBIAS, bool OUTF, bool OUTB>
__global__ __launch_bounds__(256)
void gemm_bf16(const bf16* __restrict__ A, const bf16* __restrict__ Bm,
               const float* __restrict__ bias,
               float* __restrict__ Cf, bf16* __restrict__ Cb,
               int M, int N, int K) {
    __shared__ bf16 As[128 * 32];
    __shared__ bf16 Bt[64 * 32];  // transposed [n][k]
    int t = threadIdx.x, wave = t >> 5, lane = t & 31;
    int hi = lane >> 4, ln = lane & 15;
    int m0 = blockIdx.x * 128, n0 = blockIdx.y * 64;

    unsigned as_base = lds_off(&As[0]);
    int ar = t >> 1, ac = (t & 1) * 16;                 // A-tile: 32B per thread
    unsigned as_dst = as_base + (unsigned)((ar * 32 + ac) * 2);

    v8f acc[4];
#pragma unroll
    for (int i = 0; i < 4; i++)
#pragma unroll
        for (int j = 0; j < 8; j++) acc[i][j] = 0.f;

    for (int k0 = 0; k0 < K; k0 += 32) {
        __syncthreads();
        // A tile 128x32: direct async global->LDS (row-major, no transpose)
        async_copy32(as_dst, A + (size_t)(m0 + ar) * K + k0 + ac);
        {   // B tile 32x64, stored transposed via VGPRs
            int kk = t >> 3, nc = (t & 7) * 8;
            v8bf v = *(const v8bf*)(Bm + (size_t)(k0 + kk) * N + n0 + nc);
#pragma unroll
            for (int i = 0; i < 8; i++) Bt[(nc + i) * 32 + kk] = v[i];
        }
        async_wait0();
        __syncthreads();
        v16bf a = load_frag_rows(&As[wave * 16 * 32], 32, lane);
#pragma unroll
        for (int nt = 0; nt < 4; nt++) {
            v16bf bfrag = load_frag_cols(&Bt[nt * 16 * 32], 32, lane);
            acc[nt] = __builtin_amdgcn_wmma_f32_16x16x32_bf16(
                false, a, false, bfrag, (short)0, acc[nt], false, false);
        }
    }
#pragma unroll
    for (int nt = 0; nt < 4; nt++) {
        int col = n0 + nt * 16 + ln;
        float bb = 0.f;
        if constexpr (HASBIAS) bb = bias[col];
#pragma unroll
        for (int r = 0; r < 8; r++) {
            int row = m0 + wave * 16 + r + 8 * hi;
            size_t idx = (size_t)row * N + col;
            float v = acc[nt][r] + bb;
            if constexpr (OUTF) Cf[idx] = v;
            if constexpr (OUTB) Cb[idx] = to_bf16(v);
        }
    }
}

// ---------------- Per-(b,s,h) row stats over DH=64 ----------------
__global__ void stats_kernel(const float* __restrict__ f, float* __restrict__ sn,
                             float* __restrict__ smu, float* __restrict__ svar) {
    int gid = blockIdx.x * 8 + (threadIdx.x >> 5);
    int lane = threadIdx.x & 31;
    int h = gid % H, bs = gid / H;
    const float* fr = f + (size_t)bs * D + h * DH;
    float x0 = fr[lane * 2], x1 = fr[lane * 2 + 1];
    float s = x0 + x1, q = x0 * x0 + x1 * x1;
#pragma unroll
    for (int o = 16; o > 0; o >>= 1) { s += __shfl_xor(s, o, 32); q += __shfl_xor(q, o, 32); }
    if (lane == 0) {
        float mu = s / (float)DH;
        sn[gid]  = sqrtf(q);
        smu[gid] = mu;
        svar[gid] = (q - (float)DH * mu * mu) / (float)(DH - 1);
    }
}

// ---------------- Rank-1 folding vectors u1,u2 per (h,b) ----------------
__global__ void rank1_kernel(const float* __restrict__ fv, const float* __restrict__ kmu,
                             const float* __restrict__ kvar,
                             float* __restrict__ u1, float* __restrict__ u2) {
    int hb = blockIdx.x, h = hb / Bb, b = hb % Bb;
    int t = threadIdx.x, d = t & 63, part = t >> 6;
    float a1 = 0.f, a2 = 0.f;
    for (int m = part * (S / 4); m < (part + 1) * (S / 4); m++) {
        int bs = b * S + m;
        float v = fv[(size_t)bs * D + h * DH + d];
        int sidx = bs * H + h;
        a1 += kmu[sidx] * v;
        a2 += kvar[sidx] * v;
    }
    __shared__ float l1[256], l2[256];
    l1[t] = a1; l2[t] = a2;
    __syncthreads();
    if (part == 0) {
        u1[hb * DH + d] = l1[d] + l1[d + 64] + l1[d + 128] + l1[d + 192];
        u2[hb * DH + d] = l2[d] + l2[d + 64] + l2[d + 128] + l2[d + 192];
    }
}

// ---------------- Fused attention: per (h,b), 128 query rows per block -------
__global__ __launch_bounds__(256)
void attn_kernel(const bf16* __restrict__ fq, const bf16* __restrict__ fk,
                 const bf16* __restrict__ fv,
                 const float* __restrict__ qn, const float* __restrict__ qmu,
                 const float* __restrict__ qvar, const float* __restrict__ kn,
                 const float* __restrict__ u1, const float* __restrict__ u2,
                 const float* __restrict__ cov_wp, const float* __restrict__ var_wp,
                 bf16* __restrict__ obf) {
    __shared__ bf16 kS[64 * 64];           // K tile [m][d] (async-staged)
    __shared__ bf16 vT[64 * 64];           // V tile transposed [d][m]
    __shared__ bf16 attn_s[8 * 16 * 64];   // per-wave attn strip [16][64]
    int t = threadIdx.x, wave = t >> 5, lane = t & 31;
    int hi = lane >> 4, ln = lane & 15;
    int hb = blockIdx.y, h = hb / Bb, b = hb % Bb;
    int q0 = blockIdx.x * 128 + wave * 16;

    float cov_w = cov_wp[0], var_w = var_wp[0];
    float cw3 = 1.f - cov_w - var_w;
    float c64 = cov_w * (1.f / 64.f);

    float sqn[8], smu[8], svr[8];
#pragma unroll
    for (int r = 0; r < 8; r++) {
        int sidx = (b * S + q0 + r + 8 * hi) * H + h;
        sqn[r] = qn[sidx]; smu[r] = qmu[sidx]; svr[r] = qvar[sidx];
    }
    const bf16* qbase = fq + (size_t)(b * S + q0) * D + h * DH;
    v16bf aq0 = load_frag_rows(qbase, D, lane);
    v16bf aq1 = load_frag_rows(qbase + 32, D, lane);

    v8f oacc[4];
#pragma unroll
    for (int i = 0; i < 4; i++)
#pragma unroll
        for (int j = 0; j < 8; j++) oacc[i][j] = 0.f;

    bf16* attw = &attn_s[wave * 16 * 64];
    int kr = t >> 2, kc = (t & 3) * 16;                 // K-tile: 32B per thread
    unsigned ks_dst = lds_off(&kS[0]) + (unsigned)((kr * 64 + kc) * 2);

    for (int m0 = 0; m0 < S; m0 += 64) {
        __syncthreads();
        // stage K tile via async global->LDS (row-major copy)
        async_copy32(ks_dst, fk + (size_t)(b * S + m0 + kr) * D + h * DH + kc);
#pragma unroll
        for (int i = 0; i < 16; i++) {   // stage V tile transposed (VGPR path)
            int linear = t * 16 + i;
            int ml = linear >> 6, d = linear & 63;
            vT[d * 64 + ml] = fv[(size_t)(b * S + m0 + ml) * D + h * DH + d];
        }
        async_wait0();
        __syncthreads();
#pragma unroll
        for (int ms = 0; ms < 4; ms++) {
            v8f dacc;
#pragma unroll
            for (int j = 0; j < 8; j++) dacc[j] = 0.f;
            const bf16* kb = &kS[ms * 16 * 64];
            v16bf bk0 = load_frag_cols(kb, 64, lane);
            v16bf bk1 = load_frag_cols(kb + 32, 64, lane);
            dacc = __builtin_amdgcn_wmma_f32_16x16x32_bf16(false, aq0, false, bk0, (short)0, dacc, false, false);
            dacc = __builtin_amdgcn_wmma_f32_16x16x32_bf16(false, aq1, false, bk1, (short)0, dacc, false, false);
            // elementwise: cov (dots part) + cosine scaling.
            // v_rcp_f32 (TRANS pipe, co-executes with WMMA) instead of IEEE div:
            // reference already fuzzes the denominator with +1e-8.
            float knl = kn[(b * S + m0 + ms * 16 + ln) * H + h];
#pragma unroll
            for (int r = 0; r < 8; r++) {
                float rcp = __builtin_amdgcn_rcpf(sqn[r] * knl + 1e-8f);
                float a = dacc[r] * (c64 + cw3 * rcp);
                attw[(r + 8 * hi) * 64 + ms * 16 + ln] = to_bf16(a);
            }
        }
        asm volatile("s_wait_dscnt 0" ::: "memory");  // LDS RAW (cross-lane) before frags
#pragma unroll
        for (int kc2 = 0; kc2 < 2; kc2++) {
            v16bf aa = load_frag_rows(attw + kc2 * 32, 64, lane);
#pragma unroll
            for (int dt = 0; dt < 4; dt++) {
                v16bf bv = load_frag_cols(&vT[dt * 16 * 64 + kc2 * 32], 64, lane);
                oacc[dt] = __builtin_amdgcn_wmma_f32_16x16x32_bf16(
                    false, aa, false, bv, (short)0, oacc[dt], false, false);
            }
        }
    }
    // epilogue: rank-1 terms ( -cov_w*muq*u1 + var_w/64*qvar*u2 ), store bf16
#pragma unroll
    for (int dt = 0; dt < 4; dt++) {
        int d = dt * 16 + ln;
        float uu1 = u1[hb * DH + d], uu2 = u2[hb * DH + d];
#pragma unroll
        for (int r = 0; r < 8; r++) {
            int srow = q0 + r + 8 * hi;
            float v = oacc[dt][r] - cov_w * smu[r] * uu1 + (var_w * (1.f / 64.f)) * svr[r] * uu2;
            obf[(size_t)(b * S + srow) * D + h * DH + d] = to_bf16(v);
        }
    }
}

extern "C" void kernel_launch(void* const* d_in, const int* in_sizes, int n_in,
                              void* d_out, int out_size, void* d_ws, size_t ws_size,
                              hipStream_t stream) {
    const float* q     = (const float*)d_in[0];
    const float* k     = (const float*)d_in[1];
    const float* v     = (const float*)d_in[2];
    const float* ln_g  = (const float*)d_in[3];
    const float* ln_b  = (const float*)d_in[4];
    const float* W_in  = (const float*)d_in[5];
    const float* W_out = (const float*)d_in[6];
    const float* b_out = (const float*)d_in[7];
    const float* cov_w = (const float*)d_in[8];
    const float* var_w = (const float*)d_in[9];

    const size_t N   = (size_t)Bb * S;          // 4096 tokens
    const size_t ND  = N * D;                   // 2M elems
    const size_t WW  = (size_t)D * D;           // 256K elems
    const size_t ST  = N * H;                   // 32K stats entries

    char* p = (char*)d_ws;
    bf16* Win_bf  = (bf16*)p;  p += WW * 2;
    bf16* Wout_bf = (bf16*)p;  p += WW * 2;
    bf16* xq_bf   = (bf16*)p;  p += ND * 2;
    bf16* xk_bf   = (bf16*)p;  p += ND * 2;
    bf16* xv_bf   = (bf16*)p;  p += ND * 2;
    float* fq_f   = (float*)p; p += ND * 4;
    float* fk_f   = (float*)p; p += ND * 4;
    float* fv_f   = (float*)p; p += ND * 4;
    bf16* fq_bf   = (bf16*)p;  p += ND * 2;
    bf16* fk_bf   = (bf16*)p;  p += ND * 2;
    bf16* fv_bf   = (bf16*)p;  p += ND * 2;
    float* qn_s   = (float*)p; p += ST * 4;
    float* qmu_s  = (float*)p; p += ST * 4;
    float* qvar_s = (float*)p; p += ST * 4;
    float* kn_s   = (float*)p; p += ST * 4;
    float* kmu_s  = (float*)p; p += ST * 4;
    float* kvar_s = (float*)p; p += ST * 4;
    float* u1     = (float*)p; p += (size_t)H * Bb * DH * 4;
    float* u2     = (float*)p; p += (size_t)H * Bb * DH * 4;
    bf16* o_bf    = (bf16*)p;  p += ND * 2;

    // 1) weights -> bf16
    cvt_kernel<<<(int)(WW / 256), 256, 0, stream>>>(W_in,  Win_bf,  (int)WW);
    cvt_kernel<<<(int)(WW / 256), 256, 0, stream>>>(W_out, Wout_bf, (int)WW);

    // 2) layernorm + bf16
    ln_kernel<<<(int)N, 256, 0, stream>>>(q, ln_g, ln_b, xq_bf);
    ln_kernel<<<(int)N, 256, 0, stream>>>(k, ln_g, ln_b, xk_bf);
    ln_kernel<<<(int)N, 256, 0, stream>>>(v, ln_g, ln_b, xv_bf);

    // 3) input projections (WMMA GEMM), write f32 + bf16
    dim3 ggrid((int)(N / 128), D / 64);
    gemm_bf16<false, true, true><<<ggrid, 256, 0, stream>>>(xq_bf, Win_bf, nullptr, fq_f, fq_bf, (int)N, D, D);
    gemm_bf16<false, true, true><<<ggrid, 256, 0, stream>>>(xk_bf, Win_bf, nullptr, fk_f, fk_bf, (int)N, D, D);
    gemm_bf16<false, true, true><<<ggrid, 256, 0, stream>>>(xv_bf, Win_bf, nullptr, fv_f, fv_bf, (int)N, D, D);

    // 4) per-(b,s,h) stats for q and k
    stats_kernel<<<(int)(ST / 8), 256, 0, stream>>>(fq_f, qn_s, qmu_s, qvar_s);
    stats_kernel<<<(int)(ST / 8), 256, 0, stream>>>(fk_f, kn_s, kmu_s, kvar_s);

    // 5) rank-1 folding vectors
    rank1_kernel<<<H * Bb, 256, 0, stream>>>(fv_f, kmu_s, kvar_s, u1, u2);

    // 6) fused attention (flash-style, S x S never hits HBM)
    dim3 agrid(S / 128, H * Bb);
    attn_kernel<<<agrid, 256, 0, stream>>>(fq_bf, fk_bf, fv_bf, qn_s, qmu_s, qvar_s,
                                           kn_s, u1, u2, cov_w, var_w, o_bf);

    // 7) output projection + bias -> d_out (f32)
    gemm_bf16<true, true, false><<<ggrid, 256, 0, stream>>>(o_bf, Wout_bf, b_out,
                                                            (float*)d_out, nullptr, (int)N, D, D);
}
